// EEGvae_51685636440874
// MI455X (gfx1250) — matrix operand order repaired and checked by reading, not compiled, not run on hardware
//
#include <hip/hip_runtime.h>

typedef _Float16 f16;
typedef __attribute__((ext_vector_type(16))) _Float16 v16h;
typedef __attribute__((ext_vector_type(8)))  _Float16 v8h;
typedef __attribute__((ext_vector_type(8)))  float    v8f;

#define BATCH 32768
#define WAVES 2
#define WG_SAMPLES 32
#define NWG (BATCH / WG_SAMPLES)   // 1024 workgroups for k1/k3
#define NFRAG 27                   // 18 fg1 + 3 conv1 + 6 conv2 weight fragments
#define EZERO (16 * 12 * 57)       // zeroed LDS slot for conv1 K-padding

// Native gfx1250 V_TANH_F32 if the builtin exists, else libm fallback.
__device__ __forceinline__ float fast_tanh(float x) {
#if __has_builtin(__builtin_amdgcn_tanhf)
  return __builtin_amdgcn_tanhf(x);
#elif __has_builtin(__builtin_amdgcn_tanh_f32)
  return __builtin_amdgcn_tanh_f32(x);
#else
  return tanhf(x);
#endif
}

// A-matrix (16x32 f16) element->K mapping: lane l, vector element e
__device__ __forceinline__ int kA(int e, int half) { return ((e >> 3) << 4) + half * 8 + (e & 7); }

// -------------------------------------------------------------------------
// Prep: pack all weight B-fragments into per-lane register layout
// fragW[f][lane][e] ; B(32x16 f16): element e of lane l holds K = kA(e, l>>4)
// -------------------------------------------------------------------------
__global__ __launch_bounds__(256) void eeg_prep(
    const float* __restrict__ fgw1, const float* __restrict__ c1w,
    const float* __restrict__ c2w, f16* __restrict__ fragW)
{
  int t = blockIdx.x * 256 + threadIdx.x;
  if (t >= NFRAG * 32) return;
  const int f = t >> 5, l = t & 31;
  const int m = l & 15, half = l >> 4;
  f16* dst = fragW + (size_t)t * 16;
  for (int e = 0; e < 16; ++e) {
    float v = 0.f;
    if (f < 18) {                       // fg layer1: f = nt*6 + ks  (K=168 pad 192, N=35 pad 48)
      int nt = f / 6, ks = f - nt * 6;
      int n = nt * 16 + m, k = kA(e, half) + 32 * ks;
      if (n < 35 && k < 168) v = fgw1[n * 168 + k];
    } else if (f < 21) {                // conv1: ks = f-18 (K=72 pad 96)
      int ks = f - 18;
      int k = kA(e, half) + 32 * ks;
      if (k < 72) v = c1w[m * 72 + k];
    } else {                            // conv2: f-21 = nt*3 + ks (K=96 exact)
      int g = f - 21;
      int nt = g / 3, ks = g - nt * 3;
      int k = kA(e, half) + 32 * ks;
      v = c2w[(nt * 16 + m) * 96 + k];
    }
    dst[e] = (f16)v;
  }
}

// -------------------------------------------------------------------------
// Kernel 1: fg-MLP (WMMA 168x35 + VALU 35x7) -> per-sample FIR (VALU) ->
// conv1 implicit-GEMM WMMA.  Writes pre-BN act1 (f16 [B][18][16]) + BN1 partials.
// -------------------------------------------------------------------------
__global__ __launch_bounds__(64) void eeg_k1(
    const float* __restrict__ raw, const float* __restrict__ eegf,
    const float* __restrict__ fgb1, const float* __restrict__ fgw2,
    const float* __restrict__ fgb2, const float* __restrict__ c1b,
    const f16* __restrict__ fragW,
    f16* __restrict__ act1, float* __restrict__ part1)
{
  __shared__ alignas(16) f16 eegL[WAVES][16 * 12 * 57 + 8]; // FIR output + zero slot
  __shared__ alignas(32) float ubuf[WAVES][1536];           // phase union: eegf-f16 / H / raw
  __shared__ float fL[WAVES][16 * 8];                       // per-sample 7-tap filters
  __shared__ float statL[WAVES * 2 * 32];

  const int tid  = threadIdx.x;
  const int w    = tid >> 5;
  const int l    = tid & 31;
  const int m    = l & 15;
  const int half = l >> 4;
  const int sb   = blockIdx.x * WG_SAMPLES + w * 16;
  const v16h* FW = (const v16h*)fragW;

  // ---- phase A: stage eegf tile -> LDS f16, K zero-padded 168->192 ----
  {
    f16* ef = (f16*)ubuf[w];
    const float* src = eegf + (size_t)sb * 168;     // contiguous 16x168 block
    for (int i = l; i < 16 * 168; i += 32) {
      int mm = i / 168, k = i - mm * 168;
      ef[mm * 192 + k] = (f16)src[i];
    }
    for (int i = l; i < 16 * 24; i += 32) {
      int mm = i / 24, k = i - mm * 24;
      ef[mm * 192 + 168 + k] = (f16)0.f;
    }
  }
  __syncthreads();

  // ---- build fg1 A fragments: two aligned ds_load_b128 each ----
  v16h afg[6];
  {
    const v8h* rowp = (const v8h*)((const f16*)ubuf[w] + m * 192);
#pragma unroll
    for (int ks = 0; ks < 6; ++ks) {
      v8h lo = rowp[4 * ks + half];          // K = 32ks + 8*half + [0..7]
      v8h hi = rowp[4 * ks + 2 + half];      // K = 32ks + 16 + 8*half + [0..7]
      afg[ks] = __builtin_shufflevector(lo, hi, 0,1,2,3,4,5,6,7,8,9,10,11,12,13,14,15);
    }
  }
  __syncthreads();   // ubuf reused as H below

  // ---- fg layer 1: H = tanh(eegf @ fgw1^T + b1) ----
  float* H = (float*)ubuf[w];
#pragma unroll 1
  for (int nt = 0; nt < 3; ++nt) {
    v8f acc = {};
#pragma unroll
    for (int ks = 0; ks < 6; ++ks)
      acc = __builtin_amdgcn_wmma_f32_16x16x32_f16(false, afg[ks], false,
                FW[(nt * 6 + ks) * 32 + l], (short)0, acc, false, false);
    const int n = nt * 16 + m;
    float bias = fgb1[n < 35 ? n : 0];
    if (n >= 35) bias = 0.f;                  // cols >=35 stay exactly 0
#pragma unroll
    for (int r = 0; r < 8; ++r) {
      int mm = half * 8 + r;
      H[mm * 48 + n] = fast_tanh(acc[r] + bias);
    }
  }
  __syncthreads();

  // ---- fg layer 2 (35->7) on VALU ----
  {
    float* F = fL[w];
    const int nbase = half ? 4 : 0;
    const int ncnt  = half ? 3 : 4;
    for (int i = 0; i < ncnt; ++i) {
      int n = nbase + i;
      float acc = fgb2[n];
      for (int k = 0; k < 35; ++k) acc += H[m * 48 + k] * fgw2[n * 35 + k];
      F[m * 8 + n] = fast_tanh(acc);
    }
  }
  __syncthreads();

  // ---- per-sample FIR: eeg[b,c,j] = sum_k raw[b, c*63+j+k] * filt[b,k] ----
  float* rawF = (float*)ubuf[w];              // reuse union (H dead)
  for (int c = 0; c < 12; ++c) {
    for (int i = l; i < 16 * 63; i += 32) {
      int mm = i / 63, j = i - mm * 63;
      rawF[mm * 64 + j] = raw[(size_t)(sb + mm) * 756 + c * 63 + j];
    }
    __syncthreads();
    for (int i = l; i < 16 * 57; i += 32) {
      int mm = i / 57, j = i - mm * 57;
      float acc = 0.f;
#pragma unroll
      for (int k = 0; k < 7; ++k) acc += rawF[mm * 64 + j + k] * fL[w][mm * 8 + k];
      eegL[w][(mm * 12 + c) * 57 + j] = (f16)acc;
    }
    __syncthreads();
  }
  if (l == 0) eegL[w][EZERO] = (f16)0.f;      // zero slot for K-padding
  __syncthreads();

  // ---- conv1: [B,12,57]->[B,16,18], k6 s3, implicit GEMM (K=72 pad 96) ----
  v16h bw[3];
#pragma unroll
  for (int ks = 0; ks < 3; ++ks) bw[ks] = FW[(18 + ks) * 32 + l];
  const float bias1 = c1b[m];
  float s0 = 0.f, s1 = 0.f;                   // per-lane BN stats, channel = m
  const f16* E = eegL[w];
#pragma unroll 1
  for (int t = 0; t < 18; ++t) {
    v8f acc = {};
#pragma unroll
    for (int ks = 0; ks < 3; ++ks) {
      v16h af;
#pragma unroll
      for (int e = 0; e < 16; ++e) {
        int k = kA(e, half) + 32 * ks;
        int ic = k / 6, kk = k - ic * 6;
        int idx = (k < 72) ? ((m * 12 + ic) * 57 + 3 * t + kk) : EZERO;  // branch-free pad
        af[e] = E[idx];
      }
      acc = __builtin_amdgcn_wmma_f32_16x16x32_f16(false, af, false, bw[ks],
                                                   (short)0, acc, false, false);
    }
#pragma unroll
    for (int r = 0; r < 8; ++r) {
      float v = acc[r] + bias1;
      s0 += v; s1 += v * v;
      int mm = half * 8 + r;
      act1[(((size_t)(sb + mm)) * 18 + t) * 16 + m] = (f16)v;
    }
  }

  statL[(w * 2 + 0) * 32 + l] = s0;
  statL[(w * 2 + 1) * 32 + l] = s1;
  __syncthreads();
  if (tid < 16) {
    float r0 = 0.f, r1 = 0.f;
    for (int ww = 0; ww < WAVES; ++ww) {
      r0 += statL[(ww * 2 + 0) * 32 + tid] + statL[(ww * 2 + 0) * 32 + tid + 16];
      r1 += statL[(ww * 2 + 1) * 32 + tid] + statL[(ww * 2 + 1) * 32 + tid + 16];
    }
    part1[(size_t)blockIdx.x * 32 + tid]      = r0;
    part1[(size_t)blockIdx.x * 32 + 16 + tid] = r1;
  }
}

// -------------------------------------------------------------------------
// BN finalize: reduce per-workgroup partials -> scale/shift per channel
// -------------------------------------------------------------------------
__global__ __launch_bounds__(64) void eeg_bnfin(
    const float* __restrict__ part, int nwg, int nch,
    const float* __restrict__ g, const float* __restrict__ bb,
    float* __restrict__ bn, float inv)
{
  int c = threadIdx.x;
  if (c >= nch) return;
  float s0 = 0.f, s1 = 0.f;
  for (int i = 0; i < nwg; ++i) {
    s0 += part[(size_t)i * 2 * nch + c];
    s1 += part[(size_t)i * 2 * nch + nch + c];
  }
  float mean = s0 * inv;
  float var  = s1 * inv - mean * mean;
  float sc   = g[c] * rsqrtf(var + 1e-5f);
  bn[c]       = sc;
  bn[nch + c] = bb[c] - mean * sc;
}

// -------------------------------------------------------------------------
// Kernel 3: BN1+tanh, conv2 [B,16,18]->[B,32,5] implicit GEMM (K=96 exact),
// writes pre-BN act2 (f16 [B][5][32]) + BN2 partials.
// -------------------------------------------------------------------------
__global__ __launch_bounds__(64) void eeg_k3(
    const f16* __restrict__ act1, const float* __restrict__ bn1,
    const float* __restrict__ c2b, const f16* __restrict__ fragW,
    f16* __restrict__ act2, float* __restrict__ part2)
{
  __shared__ alignas(16) f16 aL[WAVES][16 * 18 * 16];   // post-BN1 activations
  __shared__ float statL[WAVES * 2 * 2 * 32];

  const int tid  = threadIdx.x;
  const int w    = tid >> 5;
  const int l    = tid & 31;
  const int m    = l & 15;
  const int half = l >> 4;
  const int sb   = blockIdx.x * WG_SAMPLES + w * 16;
  const v16h* FW = (const v16h*)fragW;

  // stage + BN1 + tanh, packed u32 (layout [b][t][oc]; channel = idx & 15)
  // i = l + 32j  =>  channel pair (2l & 15, +1) is loop-invariant per lane
  {
    const unsigned int* src = (const unsigned int*)(act1 + (size_t)sb * 288);
    unsigned int* dst = (unsigned int*)aL[w];
    const int c0 = (2 * l) & 15;
    const float sc0 = bn1[c0],     sh0 = bn1[16 + c0];
    const float sc1 = bn1[c0 + 1], sh1 = bn1[17 + c0];
    for (int i = l; i < 16 * 288 / 2; i += 32) {
      union { unsigned int u; f16 h[2]; } p;
      p.u = src[i];
      p.h[0] = (f16)fast_tanh((float)p.h[0] * sc0 + sh0);
      p.h[1] = (f16)fast_tanh((float)p.h[1] * sc1 + sh1);
      dst[i] = p.u;
    }
  }
  __syncthreads();

  v16h bw[2][3];
#pragma unroll
  for (int nt = 0; nt < 2; ++nt)
#pragma unroll
    for (int ks = 0; ks < 3; ++ks)
      bw[nt][ks] = FW[(21 + nt * 3 + ks) * 32 + l];

  float s0[2] = {0.f, 0.f}, s1[2] = {0.f, 0.f};
  const f16* A = aL[w];
#pragma unroll 1
  for (int tt = 0; tt < 5; ++tt) {
    v16h af[3];
#pragma unroll
    for (int ks = 0; ks < 3; ++ks)
#pragma unroll
      for (int e = 0; e < 16; ++e) {
        int k = kA(e, half) + 32 * ks;          // K=96 exact, no padding
        int ic = k / 6, kk = k - ic * 6;
        af[ks][e] = A[m * 288 + (3 * tt + kk) * 16 + ic];
      }
#pragma unroll 1
    for (int nt = 0; nt < 2; ++nt) {
      v8f acc = {};
#pragma unroll
      for (int ks = 0; ks < 3; ++ks)
        acc = __builtin_amdgcn_wmma_f32_16x16x32_f16(false, af[ks], false, bw[nt][ks],
                                                     (short)0, acc, false, false);
      const int n = nt * 16 + m;
      const float bias = c2b[n];
#pragma unroll
      for (int r = 0; r < 8; ++r) {
        float v = acc[r] + bias;
        s0[nt] += v; s1[nt] += v * v;
        int mm = half * 8 + r;
        act2[(((size_t)(sb + mm)) * 5 + tt) * 32 + n] = (f16)v;
      }
    }
  }

  statL[((w * 2 + 0) * 2 + 0) * 32 + l] = s0[0];
  statL[((w * 2 + 0) * 2 + 1) * 32 + l] = s1[0];
  statL[((w * 2 + 1) * 2 + 0) * 32 + l] = s0[1];
  statL[((w * 2 + 1) * 2 + 1) * 32 + l] = s1[1];
  __syncthreads();
  if (tid < 32) {
    int nt = tid >> 4, cc = tid & 15;
    float r0 = 0.f, r1 = 0.f;
    for (int ww = 0; ww < WAVES; ++ww) {
      int b0 = ((ww * 2 + nt) * 2 + 0) * 32;
      int b1 = ((ww * 2 + nt) * 2 + 1) * 32;
      r0 += statL[b0 + cc] + statL[b0 + cc + 16];
      r1 += statL[b1 + cc] + statL[b1 + cc + 16];
    }
    part2[(size_t)blockIdx.x * 64 + tid]      = r0;
    part2[(size_t)blockIdx.x * 64 + 32 + tid] = r1;
  }
}

// -------------------------------------------------------------------------
// Kernel 5: BN2+tanh, heads, reparameterization, decoder.
// -------------------------------------------------------------------------
__global__ __launch_bounds__(256) void eeg_dec(
    const f16* __restrict__ act2, const float* __restrict__ bn2,
    const float* __restrict__ eps, const float* __restrict__ muw,
    const float* __restrict__ mub, const float* __restrict__ lvw,
    const float* __restrict__ lvb, const float* __restrict__ dw1,
    const float* __restrict__ db1, const float* __restrict__ dw2,
    const float* __restrict__ db2, float* __restrict__ out)
{
  const int b = blockIdx.x * 256 + threadIdx.x;
  const unsigned int* row = (const unsigned int*)(act2 + (size_t)b * 160); // [tt][oc]
  float m0 = mub[0], m1 = mub[1], v0 = lvb[0], v1 = lvb[1];
  for (int ocp = 0; ocp < 16; ++ocp) {          // oc pair (2*ocp, 2*ocp+1)
    int oc0 = 2 * ocp;
    float sc0 = bn2[oc0], sh0 = bn2[32 + oc0];
    float sc1 = bn2[oc0 + 1], sh1 = bn2[33 + oc0];
#pragma unroll
    for (int tt = 0; tt < 5; ++tt) {
      union { unsigned int u; f16 h[2]; } p;
      p.u = row[tt * 16 + ocp];
      float e0 = fast_tanh((float)p.h[0] * sc0 + sh0);
      float e1 = fast_tanh((float)p.h[1] * sc1 + sh1);
      int j0 = oc0 * 5 + tt, j1 = j0 + 5;       // en index = oc*5 + tt
      m0 += e0 * muw[j0] + e1 * muw[j1];
      m1 += e0 * muw[160 + j0] + e1 * muw[160 + j1];
      v0 += e0 * lvw[j0] + e1 * lvw[j1];
      v1 += e0 * lvw[160 + j0] + e1 * lvw[160 + j1];
    }
  }
  float z0 = m0 + eps[(size_t)b * 2 + 0] * expf(0.5f * v0);
  float z1 = m1 + eps[(size_t)b * 2 + 1] * expf(0.5f * v1);
  float d1[10];
#pragma unroll
  for (int i = 0; i < 10; ++i)
    d1[i] = fast_tanh(db1[i] + z0 * dw1[i * 2 + 0] + z1 * dw1[i * 2 + 1]);
  float* orow = out + (size_t)b * 50;
#pragma unroll 1
  for (int oo = 0; oo < 50; ++oo) {
    float acc = db2[oo];
#pragma unroll
    for (int i = 0; i < 10; ++i) acc += d1[i] * dw2[oo * 10 + i];
    orow[oo] = fast_tanh(acc);
  }
  out[(size_t)BATCH * 50 + (size_t)b * 2 + 0] = m0;
  out[(size_t)BATCH * 50 + (size_t)b * 2 + 1] = m1;
  out[(size_t)BATCH * 52 + (size_t)b * 2 + 0] = v0;
  out[(size_t)BATCH * 52 + (size_t)b * 2 + 1] = v1;
}

extern "C" void kernel_launch(void* const* d_in, const int* in_sizes, int n_in,
                              void* d_out, int out_size, void* d_ws, size_t ws_size,
                              hipStream_t stream)
{
  (void)in_sizes; (void)n_in; (void)out_size; (void)ws_size;
  const float* raw  = (const float*)d_in[0];
  const float* eegf = (const float*)d_in[1];
  const float* eps  = (const float*)d_in[2];
  const float* fgw1 = (const float*)d_in[3];
  const float* fgb1 = (const float*)d_in[4];
  const float* fgw2 = (const float*)d_in[5];
  const float* fgb2 = (const float*)d_in[6];
  const float* c1w  = (const float*)d_in[7];
  const float* c1b  = (const float*)d_in[8];
  const float* bn1g = (const float*)d_in[9];
  const float* bn1b = (const float*)d_in[10];
  const float* c2w  = (const float*)d_in[11];
  const float* c2b  = (const float*)d_in[12];
  const float* bn2g = (const float*)d_in[13];
  const float* bn2b = (const float*)d_in[14];
  const float* muw  = (const float*)d_in[15];
  const float* mub  = (const float*)d_in[16];
  const float* lvw  = (const float*)d_in[17];
  const float* lvb  = (const float*)d_in[18];
  const float* dw1  = (const float*)d_in[19];
  const float* db1  = (const float*)d_in[20];
  const float* dw2  = (const float*)d_in[21];
  const float* db2  = (const float*)d_in[22];
  float* out = (float*)d_out;

  char* ws = (char*)d_ws;
  size_t o = 0;
  f16*   act1  = (f16*)(ws + o);   o += (size_t)BATCH * 288 * sizeof(f16);
  f16*   act2  = (f16*)(ws + o);   o += (size_t)BATCH * 160 * sizeof(f16);
  float* part1 = (float*)(ws + o); o += (size_t)NWG * 32 * sizeof(float);
  float* part2 = (float*)(ws + o); o += (size_t)NWG * 64 * sizeof(float);
  f16*   fragW = (f16*)(ws + o);   o += (size_t)NFRAG * 32 * 16 * sizeof(f16);
  float* bn1   = (float*)(ws + o); o += 32 * sizeof(float);
  float* bn2   = (float*)(ws + o); o += 64 * sizeof(float);

  eeg_prep<<<(NFRAG * 32 + 255) / 256, 256, 0, stream>>>(fgw1, c1w, c2w, fragW);
  eeg_k1<<<NWG, 64, 0, stream>>>(raw, eegf, fgb1, fgw2, fgb2, c1b, fragW, act1, part1);
  eeg_bnfin<<<1, 64, 0, stream>>>(part1, NWG, 16, bn1g, bn1b, bn1,
                                  1.f / (float)((size_t)BATCH * 18));
  eeg_k3<<<NWG, 64, 0, stream>>>(act1, bn1, c2b, fragW, act2, part2);
  eeg_bnfin<<<1, 64, 0, stream>>>(part2, NWG, 32, bn2g, bn2b, bn2,
                                  1.f / (float)((size_t)BATCH * 5));
  eeg_dec<<<BATCH / 256, 256, 0, stream>>>(act2, bn2, eps, muw, mub, lvw, lvb,
                                           dw1, db1, dw2, db2, out);
}
